// VQVAE_23811298689888
// MI455X (gfx1250) — matrix operand order repaired and checked by reading, not compiled, not run on hardware
//
#include <hip/hip_runtime.h>

typedef __attribute__((ext_vector_type(16))) __bf16 v16bf;
typedef __attribute__((ext_vector_type(8)))  float  v8f;
typedef __attribute__((ext_vector_type(4)))  unsigned int v4u;
typedef __attribute__((ext_vector_type(8)))  int    v8i;
typedef __attribute__((ext_vector_type(4)))  int    v4i;

typedef unsigned short u16;
typedef unsigned int   u32;

#if defined(__AMDGCN__) && __has_builtin(__builtin_amdgcn_tensor_load_to_lds)
#define USE_TDM 1
#else
#define USE_TDM 0
#endif

__device__ __forceinline__ u16 f2bf(float f) {
    u32 u = __float_as_uint(f);
    u32 r = u + 0x7FFFu + ((u >> 16) & 1u);   // round-to-nearest-even
    return (u16)(r >> 16);
}
__device__ __forceinline__ float bf2f(u16 h) {
    return __uint_as_float(((u32)h) << 16);
}

union Frag {
    v16bf v;
    uint4 q[2];
};

// ---------------------------------------------------------------------------
// Implicit-GEMM conv, bf16 operands, f32 accum (v_wmma_f32_16x16x32_bf16).
// out[M=Cout][cols] = W[M][K] * im2col(X)[K][cols],  K tap-major: tap*Cin+ci
// parity mode: convT(k4,s2,p1) split into 4 parity-class GEMMs (K=Cin*4) on the
// half-res output grid; epilogue scatters into the full-res canonical layout.
// Block 256 thr = 8 waves; tile M128 x N128; waves 4(M) x 2(N) -> 2x4 WMMA tiles.
// A tile DMA'd by the Tensor Data Mover (wave 0, TENSORcnt), double-buffered LDS.
// ---------------------------------------------------------------------------
struct ConvDesc {
    const u16* W;       // bf16 [M][K]
    const u16* X;       // bf16 [Cin][NB*Hin*Win]
    int M, K, Cin, ks, stride, pad;
    int NB, Hin, Win;
    int hwShift, wShift, cinShift;   // output-grid shifts (half-res in parity mode)
    int parity, phH, phW;            // parity-class convT mode
    int in_relu, out_relu;
    int has_bn, has_bias, has_resid, logits;
    const float *g, *b, *m, *v, *bias;
    const u16* resid;   // bf16 [M][colsOut] (may alias outb)
    u16*   outb;        // bf16 [M][colsOut] or null
    float* outf;        // fp32; logits==1 -> permuted (B,3,H,W,256) store
};

#define MT  128
#define NT  128
#define KT  32
#define LDP 40   // padded LDS row pitch (bf16): 80B -> conflict-free b128 frag reads

#if USE_TDM
// TDM descriptor for a [128 x 32] bf16 tile of row-major [M x K] weights, with
// 16B LDS padding after every 64B row (-> LDP=40) and zero-fill past K.
__device__ __forceinline__ void tdm_load_a(const ConvDesc& d, int m0, int k0,
                                           u32 ldsOff) {
    unsigned long long ga =
        (unsigned long long)(uintptr_t)(d.W + (size_t)m0 * d.K + k0);
    u32 td0 = (u32)(d.K - k0);         // remaining K columns (OOB reads -> 0)
    u32 td1 = (u32)(d.M - m0);
    u32 st0 = (u32)d.K;                // row stride, data_size units
    v4u g0 = { 1u,                     // count=1, user descriptor
               ldsOff,                 // lds_addr
               (u32)ga,
               (u32)((ga >> 32) & 0x01FFFFFFu) | (2u << 30) };  // addr hi | type=2
    v8i g1;
    g1[0] = (int)((1u << 16)           // data_size = 2B
                | (1u << 20)           // pad_enable
                | (3u << 22)           // pad_interval: 16 dwords (64B)
                | (3u << 25));         // pad_amount:   4 dwords (16B)
    g1[1] = (int)(td0 << 16);                          // tensor_dim0 lo16
    g1[2] = (int)((td0 >> 16) | (td1 << 16));          // td0 hi16 | td1 lo16
    g1[3] = (int)((td1 >> 16) | (32u << 16));          // td1 hi16 | tile_dim0=32
    g1[4] = (int)128;                                  // tile_dim1=128, tile_dim2=0
    g1[5] = (int)st0;                                  // dim0 stride lo32
    g1[6] = 0;                                         // stride hi16 | dim1 stride
    g1[7] = 0;
    v4i gz = {0, 0, 0, 0};
#if __clang_major__ >= 23
    v8i gz8 = {0, 0, 0, 0, 0, 0, 0, 0};
    __builtin_amdgcn_tensor_load_to_lds(g0, g1, gz, gz, gz8, 0);
#else
    __builtin_amdgcn_tensor_load_to_lds(g0, g1, gz, gz, 0);
#endif
}
#endif

__device__ __forceinline__ void load_a_tile(const ConvDesc& d, int m0, int k0,
                                            u16* AsBuf, int t) {
    const int row = t >> 1;
    const int kh0 = (t & 1) * 16;
    const u16* src = d.W + (size_t)(m0 + row) * d.K + k0 + kh0;
    u16* dst = AsBuf + row * LDP + kh0;
#pragma unroll
    for (int i = 0; i < 8; i++) {      // K always even: safe pair loads
        int kkp = k0 + kh0 + 2 * i;
        u32 v = (kkp < d.K) ? *(const u32*)(src + 2 * i) : 0u;
        *(u32*)(dst + 2 * i) = v;
    }
}

// B tile (im2col gather). Thread -> one K index x 16 consecutive cols, so global
// traffic is coalesced (contiguous per fixed K in [C][cols] layout); 1x1 convs
// take a 2x b128 vector path. LDS stored [col][k] with XOR chunk swizzle.
__device__ __forceinline__ void load_b_tile(const ConvDesc& d, int n0, int k0,
                                            u16* BsBuf, int t, int cols, int inHW) {
    const int kk_l = t >> 3;
    const int col0 = (t & 7) * 16;
    const int kk   = k0 + kk_l;
    const int pos  = (((kk_l >> 3) ^ (t & 3)) << 3) | (kk_l & 7);   // swizzled k slot
    union { uint4 q[2]; u16 u[16]; } tmp;

    if (d.ks == 1 && !d.parity && !d.in_relu && kk < d.K) {
        const uint4* src = (const uint4*)(d.X + (size_t)kk * cols + n0 + col0);
        tmp.q[0] = src[0];
        tmp.q[1] = src[1];
    } else if (d.parity) {
        // convT parity class: 2x2 valid taps, half-res output grid
        const int tap = kk >> d.cinShift;          // 0..3
        const int ci  = kk & (d.Cin - 1);
        const int fh2 = tap >> 1, fw2 = tap & 1;
#pragma unroll
        for (int i = 0; i < 16; i++) {
            int ng  = n0 + col0 + i;
            int bb  = ng >> d.hwShift;
            int rem = ng & ((1 << d.hwShift) - 1);
            int ho2 = rem >> d.wShift;
            int wo2 = rem & ((1 << d.wShift) - 1);
            int hi  = ho2 - fh2 + d.phH;
            int wi  = wo2 - fw2 + d.phW;
            u16 val = 0;
            if ((hi >= 0) & (hi < d.Hin) & (wi >= 0) & (wi < d.Win))
                val = d.X[(size_t)ci * (d.NB * inHW) + (size_t)bb * inHW
                          + hi * d.Win + wi];
            tmp.u[i] = val;
        }
    } else {
        int tap, ci;
        if (d.ks == 1)            { tap = 0;                ci = kk; }
        else if (d.cinShift >= 0) { tap = kk >> d.cinShift; ci = kk & (d.Cin - 1); }
        else                      { ci = kk % d.Cin;        tap = kk / d.Cin; }
        int fh, fw;
        if (d.ks == 4)      { fh = tap >> 2;         fw = tap & 3; }
        else if (d.ks == 3) { fh = (tap * 11) >> 5;  fw = tap - fh * 3; } // tap<9
        else                { fh = 0;                fw = 0; }
        const bool kv = (kk < d.K);
#pragma unroll
        for (int i = 0; i < 16; i++) {
            int ng  = n0 + col0 + i;
            int bb  = ng >> d.hwShift;
            int rem = ng & ((1 << d.hwShift) - 1);
            int ho  = rem >> d.wShift;
            int wo  = rem & ((1 << d.wShift) - 1);
            u16 val = 0;
            if (kv) {
                int hi = ho * d.stride - d.pad + fh;
                int wi = wo * d.stride - d.pad + fw;
                if ((hi >= 0) & (hi < d.Hin) & (wi >= 0) & (wi < d.Win)) {
                    val = d.X[(size_t)ci * (d.NB * inHW) + (size_t)bb * inHW
                              + hi * d.Win + wi];
                    if (d.in_relu && (val & 0x8000u)) val = 0;   // bf16 relu
                }
            }
            tmp.u[i] = val;
        }
    }
#pragma unroll
    for (int i = 0; i < 16; i++)
        BsBuf[(col0 + i) * LDP + pos] = tmp.u[i];
}

__global__ __launch_bounds__(256) void conv_wmma(ConvDesc d) {
    __shared__ __align__(16) u16 As[2][MT * LDP];
    __shared__ __align__(16) u16 Bs[2][NT * LDP];

    const int hw   = 1 << d.hwShift;           // output-grid HW (half-res if parity)
    const int cols = d.NB << d.hwShift;
    const int inHW = d.Hin * d.Win;
    const int t    = threadIdx.x;
    const int lane = t & 31;
    const int wid  = t >> 5;
    const int wm   = wid & 3;
    const int wn   = wid >> 2;
    const int m0   = blockIdx.y * MT;
    const int n0   = blockIdx.x * NT;
    const int hl   = lane >> 4;
    const int lr   = lane & 15;

    const v8f zero = {0.f, 0.f, 0.f, 0.f, 0.f, 0.f, 0.f, 0.f};
    v8f acc[2][4];
#pragma unroll
    for (int i = 0; i < 2; i++)
#pragma unroll
        for (int j = 0; j < 4; j++) acc[i][j] = zero;

    const int nSteps = (d.K + KT - 1) / KT;

    // ---- prologue: stage K-step 0 into buffer 0
#if USE_TDM
    if (wid == 0) tdm_load_a(d, m0, 0, (u32)(uintptr_t)&As[0][0]);
#else
    load_a_tile(d, m0, 0, &As[0][0], t);
#endif
    load_b_tile(d, n0, 0, &Bs[0][0], t, cols, inHW);
#if USE_TDM
    if (wid == 0) __builtin_amdgcn_s_wait_tensorcnt(0);
#endif
    __syncthreads();

    for (int s = 0; s < nSteps; s++) {
        const int cur = s & 1, nxt = cur ^ 1;
        if (s + 1 < nSteps) {          // prefetch next K-step into other buffer
#if USE_TDM
            if (wid == 0) tdm_load_a(d, m0, (s + 1) * KT, (u32)(uintptr_t)&As[nxt][0]);
#else
            load_a_tile(d, m0, (s + 1) * KT, &As[nxt][0], t);
#endif
            load_b_tile(d, n0, (s + 1) * KT, &Bs[nxt][0], t, cols, inHW);
        }
        // ---- fragments (contiguous k = 16*hl + s in both A and B) + 8 WMMAs on current buffer
        Frag a[2], bf[4];
        const int ab = hl * 16;
#pragma unroll
        for (int i = 0; i < 2; i++) {
            const u16* p = &As[cur][(wm * 32 + i * 16 + lr) * LDP];
            a[i].q[0] = *(const uint4*)(p + ab);
            a[i].q[1] = *(const uint4*)(p + ab + 8);
        }
#pragma unroll
        for (int j = 0; j < 4; j++) {
            const int swz = (wn * 4 + j) & 3;          // matches writer swizzle
            const u16* p = &Bs[cur][(wn * 64 + j * 16 + lr) * LDP];
            bf[j].q[0] = *(const uint4*)(p + (((hl * 2) ^ swz) << 3));
            bf[j].q[1] = *(const uint4*)(p + (((hl * 2 + 1) ^ swz) << 3));
        }
#pragma unroll
        for (int i = 0; i < 2; i++)
#pragma unroll
            for (int j = 0; j < 4; j++)
                acc[i][j] = __builtin_amdgcn_wmma_f32_16x16x32_bf16(
                    false, a[i].v, false, bf[j].v, (short)0, acc[i][j], false, false);
#if USE_TDM
        if (wid == 0) __builtin_amdgcn_s_wait_tensorcnt(0);
#endif
        __syncthreads();
    }

    // ---- epilogue: per-channel scale/shift hoisted (bias/BN folded)
    const int colsOut = d.parity ? (cols << 2) : cols;
    float scl[2][8], sft[2][8];
#pragma unroll
    for (int i = 0; i < 2; i++)
#pragma unroll
        for (int r = 0; r < 8; r++) {
            int c = m0 + wm * 32 + i * 16 + r + hl * 8;
            float sc = 1.f, sh = 0.f;
            if (d.has_bias) sh = d.bias[c];
            if (d.has_bn) {
                float s_ = d.g[c] / sqrtf(d.v[c] + 1e-5f);
                sh = (sh - d.m[c]) * s_ + d.b[c];
                sc = s_;
            }
            scl[i][r] = sc; sft[i][r] = sh;
        }
#pragma unroll
    for (int i = 0; i < 2; i++) {
#pragma unroll
        for (int j = 0; j < 4; j++) {
            const int ng = n0 + wn * 64 + j * 16 + lr;
            int ngS = ng;
            if (d.parity) {    // scatter half-res col -> full-res canonical col
                int bb  = ng >> d.hwShift, rem = ng & (hw - 1);
                int ho2 = rem >> d.wShift;
                int wo2 = rem & ((1 << d.wShift) - 1);
                ngS = (bb << (d.hwShift + 2))
                    + (((ho2 << 1) + d.phH) << (d.wShift + 1))
                    + (wo2 << 1) + d.phW;
            }
#pragma unroll
            for (int r = 0; r < 8; r++) {
                const int c = m0 + wm * 32 + i * 16 + r + hl * 8;
                float val = acc[i][j][r] * scl[i][r] + sft[i][r];
                if (d.has_resid) val += bf2f(d.resid[(size_t)c * colsOut + ngS]);
                if (d.out_relu) val = fmaxf(val, 0.f);
                if (d.outb) d.outb[(size_t)c * colsOut + ngS] = f2bf(val);
                if (d.outf) {
                    if (d.logits) {   // (B, 3, H, W, 256)
                        int bb = ng >> d.hwShift, rem = ng & (hw - 1);
                        size_t oi = (((size_t)bb * 3 + (c >> 8)) * hw + rem) * 256
                                    + (c & 255);
                        d.outf[oi] = val;
                    } else {
                        d.outf[(size_t)c * colsOut + ngS] = val;
                    }
                }
            }
        }
    }
}

// ---------------------------------------------------------------------------
// Codebook: dist[m][col] = |e_m|^2 - 2 e_m . z_col via WMMA, fused argmin.
// ---------------------------------------------------------------------------
__global__ __launch_bounds__(32) void codebook_argmin(
        const u16* embB, const float* e2, const u16* zact,
        int* idx, int* hist, int cols) {
    __shared__ __align__(16) u16 zt[16 * 72];
    const int nh   = blockIdx.y;
    const int c0   = blockIdx.x * 16;
    const int lane = threadIdx.x;
    for (int i = lane; i < 16 * 64; i += 32) {
        int cl = i >> 6, dd = i & 63;
        zt[cl * 72 + dd] = zact[(size_t)(nh * 64 + dd) * cols + c0 + cl];
    }
    __syncthreads();
    const int hl = lane >> 4, lr = lane & 15;

    Frag b0, b1;
    {
        const u16* p = &zt[lr * 72];
        int s0 = hl * 16;
        b0.q[0] = *(const uint4*)(p + s0);
        b0.q[1] = *(const uint4*)(p + s0 + 8);
        int s1 = 32 + hl * 16;
        b1.q[0] = *(const uint4*)(p + s1);
        b1.q[1] = *(const uint4*)(p + s1 + 8);
    }
    float best = 3.4e38f;
    int   bi   = 0;
    const u16* eb = embB + (size_t)nh * 512 * 64;
    for (int mt = 0; mt < 32; mt++) {
        Frag a0, a1;
        const u16* p = eb + (size_t)(mt * 16 + lr) * 64;
        int s0 = hl * 16;
        a0.q[0] = *(const uint4*)(p + s0);
        a0.q[1] = *(const uint4*)(p + s0 + 8);
        int s1 = 32 + hl * 16;
        a1.q[0] = *(const uint4*)(p + s1);
        a1.q[1] = *(const uint4*)(p + s1 + 8);
        v8f acc = {0.f, 0.f, 0.f, 0.f, 0.f, 0.f, 0.f, 0.f};
        acc = __builtin_amdgcn_wmma_f32_16x16x32_bf16(false, a0.v, false, b0.v,
                                                      (short)0, acc, false, false);
        acc = __builtin_amdgcn_wmma_f32_16x16x32_bf16(false, a1.v, false, b1.v,
                                                      (short)0, acc, false, false);
#pragma unroll
        for (int r = 0; r < 8; r++) {
            int   m = mt * 16 + r + hl * 8;
            float dist = e2[nh * 512 + m] - 2.f * acc[r];
            if (dist < best) { best = dist; bi = m; }
        }
    }
    float ob = __shfl_xor(best, 16);
    int   oi = __shfl_xor(bi, 16);
    if (ob < best || (ob == best && oi < bi)) { best = ob; bi = oi; }
    if (lane < 16) {
        idx[nh * cols + c0 + lr] = bi;
        atomicAdd(&hist[nh * 512 + bi], 1);
    }
}

// ---------------------------------------------------------------------------
// Helpers
// ---------------------------------------------------------------------------
__global__ void f32_to_bf16(const float* s, u16* d, int n) {
    int i = blockIdx.x * blockDim.x + threadIdx.x;
    if (i < n) d[i] = f2bf(s[i]);
}
// conv weights OIHW -> bf16 [Cout][tap*Cin + ci], tap = fh*ks+fw
__global__ void conv_w_to_bf16(const float* w, u16* dst, int Cin, int Cout, int ksq) {
    int i = blockIdx.x * blockDim.x + threadIdx.x;
    if (i >= Cout * Cin * ksq) return;
    int tap = i % ksq;
    int q   = i / ksq;
    int ci  = q % Cin;
    int co  = q / Cin;
    dst[(size_t)co * (Cin * ksq) + tap * Cin + ci] = f2bf(w[i]);
}
// convT weights (Cin,Cout,4,4) -> 4 parity-packed bf16 matrices
// dst[ph*2+pw][Cout][(fh2*2+fw2)*Cin + ci]
__global__ void convt_w_parity(const float* w, u16* dst, int Cin, int Cout) {
    int i = blockIdx.x * blockDim.x + threadIdx.x;
    if (i >= Cin * Cout * 16) return;
    int f   = i & 15;
    int q   = i >> 4;
    int co  = q % Cout;
    int ci  = q / Cout;
    int fh  = f >> 2, fw = f & 3;
    int ph  = 1 - (fh & 1), pw = 1 - (fw & 1);
    int fh2 = fh >> 1,      fw2 = fw >> 1;
    size_t o = (((size_t)(ph * 2 + pw) * Cout + co) * (Cin * 4))
             + (fh2 * 2 + fw2) * Cin + ci;
    dst[o] = f2bf(w[i]);
}
// x (N,C,H,W) fp32 -> bf16 [C][N*H*W]   (HW = 4096)
__global__ void x_to_act(const float* x, u16* d, int NB, int C) {
    int i = blockIdx.x * blockDim.x + threadIdx.x;
    if (i >= NB * C * 4096) return;
    int p = i & 4095;
    int t = i >> 12;
    int c = t % 3;
    int b = t / 3;
    d[(size_t)c * NB * 4096 + b * 4096 + p] = f2bf(x[i]);
}
__global__ void emb_norms(const float* emb, float* e2, int NM, int D) {
    int i = blockIdx.x * blockDim.x + threadIdx.x;
    if (i >= NM) return;
    float s = 0.f;
    for (int d = 0; d < D; d++) { float e = emb[i * D + d]; s += e * e; }
    e2[i] = s;
}
__global__ void init_ws(int* hist, int n) {
    int i = blockIdx.x * blockDim.x + threadIdx.x;
    if (i < n) hist[i] = 0;
}
__global__ void quantize_gather(const float* emb, const int* idx, const u16* zact,
                                u16* qact, float* partial) {
    int gid = blockIdx.x * blockDim.x + threadIdx.x;   // 512 * 4096 exactly
    int col = gid & 4095;
    int c   = gid >> 12;
    int nh  = c >> 6;
    int dd  = c & 63;
    int id  = idx[nh * 4096 + col];
    float q = emb[((size_t)nh * 512 + id) * 64 + dd];
    qact[(size_t)c * 4096 + col] = f2bf(q);
    float z  = bf2f(zact[(size_t)c * 4096 + col]);
    float df = z - q;
    float d2 = df * df;
    __shared__ float red[256];
    red[threadIdx.x] = d2;
    __syncthreads();
    for (int s = 128; s > 0; s >>= 1) {
        if (threadIdx.x < s) red[threadIdx.x] += red[threadIdx.x + s];
        __syncthreads();
    }
    if (threadIdx.x == 0) partial[blockIdx.x] = red[0];
}
__global__ void finalize(const int* hist, const float* partial, int nPart,
                         float* out, size_t logits_elems) {
    if (threadIdx.x != 0 || blockIdx.x != 0) return;
    float lsum = 0.f;
    for (int i = 0; i < nPart; i++) lsum += partial[i];
    float perp = 0.f;
    for (int nh = 0; nh < 8; nh++) {
        float ent = 0.f;
        for (int m = 0; m < 512; m++) {
            float p = (float)hist[nh * 512 + m] * (1.f / 4096.f);
            ent += p * logf(p + 1e-10f);
        }
        perp += expf(-ent);
    }
    out[logits_elems]     = 0.25f * lsum / 2097152.0f;
    out[logits_elems + 1] = perp;
}

// ---------------------------------------------------------------------------
extern "C" void kernel_launch(void* const* d_in, const int* in_sizes, int n_in,
                              void* d_out, int out_size, void* d_ws, size_t ws_size,
                              hipStream_t stream) {
    (void)in_sizes; (void)n_in; (void)out_size; (void)ws_size;
    enum {
        IN_X = 0,
        E_W1 = 1, E_BN1 = 2, E_W2 = 6, E_BN2 = 7,
        E_R1_W1 = 11, E_R1_BN1 = 12, E_R1_W2 = 16, E_R1_BN2 = 17,
        E_R2_W1 = 21, E_R2_BN1 = 22, E_R2_W2 = 26, E_R2_BN2 = 27,
        E_W3 = 31, E_B3 = 32, EMB = 33,
        D_W1 = 34, D_BN1 = 35,
        D_R1_W1 = 39, D_R1_BN1 = 40, D_R1_W2 = 44, D_R1_BN2 = 45,
        D_R2_W1 = 49, D_R2_BN1 = 50, D_R2_W2 = 54, D_R2_BN2 = 55,
        D_WT1 = 59, D_BN2X = 60, D_WT2 = 64, D_BN3 = 65,
        D_W4 = 69, D_B4 = 70
    };
    auto FP = [&](int i) { return (const float*)d_in[i]; };

    char*  base = (char*)d_ws;
    size_t off  = 0;
    auto alloc = [&](size_t bytes) -> char* {
        off = (off + 255) & ~(size_t)255;
        char* p = base + off;
        off += bytes;
        return p;
    };
    const int NB = 16;
    u16* wl0   = (u16*)alloc((size_t)256 * 48 * 2);
    u16* wl1   = (u16*)alloc((size_t)256 * 4096 * 2);
    u16* wer1a = (u16*)alloc((size_t)256 * 2304 * 2);
    u16* wer1b = (u16*)alloc((size_t)256 * 256 * 2);
    u16* wer2a = (u16*)alloc((size_t)256 * 2304 * 2);
    u16* wer2b = (u16*)alloc((size_t)256 * 256 * 2);
    u16* wl4   = (u16*)alloc((size_t)512 * 256 * 2);
    u16* embB  = (u16*)alloc((size_t)8 * 512 * 64 * 2);
    u16* wd1   = (u16*)alloc((size_t)256 * 512 * 2);
    u16* wdr1a = (u16*)alloc((size_t)256 * 2304 * 2);
    u16* wdr1b = (u16*)alloc((size_t)256 * 256 * 2);
    u16* wdr2a = (u16*)alloc((size_t)256 * 2304 * 2);
    u16* wdr2b = (u16*)alloc((size_t)256 * 256 * 2);
    u16* wt1   = (u16*)alloc((size_t)4 * 256 * 1024 * 2);   // parity-packed
    u16* wt2   = (u16*)alloc((size_t)4 * 256 * 1024 * 2);
    u16* wd4   = (u16*)alloc((size_t)768 * 256 * 2);
    u16* xact  = (u16*)alloc((size_t)3 * 65536 * 2);
    u16* h1    = (u16*)alloc((size_t)256 * 16384 * 2);
    u16* h2    = (u16*)alloc((size_t)256 * 4096 * 2);
    u16* tb    = (u16*)alloc((size_t)256 * 4096 * 2);
    u16* z     = (u16*)alloc((size_t)512 * 4096 * 2);
    u16* qv    = (u16*)alloc((size_t)512 * 4096 * 2);
    u16* dd1   = (u16*)alloc((size_t)256 * 4096 * 2);
    u16* u1    = (u16*)alloc((size_t)256 * 16384 * 2);
    u16* u2    = (u16*)alloc((size_t)256 * 65536 * 2);
    float* e2   = (float*)alloc((size_t)4096 * 4);
    int*   idx  = (int*)alloc((size_t)8 * 4096 * 4);
    int*   hist = (int*)alloc((size_t)4096 * 4);
    float* part = (float*)alloc((size_t)8192 * 4);

    auto cvt = [&](int ii, u16* dst, int n) {                 // 1x1: K order = ci
        f32_to_bf16<<<(n + 255) / 256, 256, 0, stream>>>(FP(ii), dst, n);
    };
    auto cvtk = [&](int ii, u16* dst, int Cin, int Cout, int ksq) {  // tap-major
        int n = Cin * Cout * ksq;
        conv_w_to_bf16<<<(n + 255) / 256, 256, 0, stream>>>(FP(ii), dst, Cin, Cout, ksq);
    };
    cvtk(E_W1, wl0, 3, 256, 16);
    cvtk(E_W2, wl1, 256, 256, 16);
    cvtk(E_R1_W1, wer1a, 256, 256, 9);   cvt(E_R1_W2, wer1b, 256 * 256);
    cvtk(E_R2_W1, wer2a, 256, 256, 9);   cvt(E_R2_W2, wer2b, 256 * 256);
    cvt(E_W3, wl4, 512 * 256);
    cvt(EMB, embB, 8 * 512 * 64);
    cvt(D_W1, wd1, 256 * 512);
    cvtk(D_R1_W1, wdr1a, 256, 256, 9);   cvt(D_R1_W2, wdr1b, 256 * 256);
    cvtk(D_R2_W1, wdr2a, 256, 256, 9);   cvt(D_R2_W2, wdr2b, 256 * 256);
    convt_w_parity<<<(256 * 256 * 16 + 255) / 256, 256, 0, stream>>>(FP(D_WT1), wt1, 256, 256);
    convt_w_parity<<<(256 * 256 * 16 + 255) / 256, 256, 0, stream>>>(FP(D_WT2), wt2, 256, 256);
    cvt(D_W4, wd4, 768 * 256);
    x_to_act<<<(NB * 3 * 4096 + 255) / 256, 256, 0, stream>>>(FP(IN_X), xact, NB, 3);
    emb_norms<<<(4096 + 255) / 256, 256, 0, stream>>>(FP(EMB), e2, 4096, 64);
    init_ws<<<16, 256, 0, stream>>>(hist, 4096);

    auto conv = [&](const u16* W, const u16* X, int M, int K, int Cin, int ks,
                    int st, int pd, int Hin, int Hout, int in_relu,
                    int bnIdx, int biasIdx, const u16* resid, int out_relu,
                    u16* outb, float* outf, int logits) {
        ConvDesc cd{};
        cd.W = W; cd.X = X; cd.M = M; cd.K = K; cd.Cin = Cin; cd.ks = ks;
        cd.stride = st; cd.pad = pd; cd.NB = NB;
        cd.Hin = Hin; cd.Win = Hin;
        int wsh = __builtin_ctz(Hout);
        cd.wShift = wsh; cd.hwShift = 2 * wsh;
        cd.cinShift = (Cin & (Cin - 1)) ? -1 : __builtin_ctz(Cin);
        cd.parity = 0; cd.phH = 0; cd.phW = 0;
        cd.in_relu = in_relu; cd.out_relu = out_relu;
        cd.has_bn = (bnIdx >= 0);
        if (cd.has_bn) { cd.g = FP(bnIdx); cd.b = FP(bnIdx + 1);
                         cd.m = FP(bnIdx + 2); cd.v = FP(bnIdx + 3); }
        cd.has_bias = (biasIdx >= 0);
        if (cd.has_bias) cd.bias = FP(biasIdx);
        cd.has_resid = (resid != nullptr); cd.resid = resid;
        cd.outb = outb; cd.outf = outf; cd.logits = logits;
        int cols = NB * Hout * Hout;
        dim3 grid(cols / NT, M / MT);
        conv_wmma<<<grid, 256, 0, stream>>>(cd);
    };
    // convT(k4,s2,p1) as 4 parity-class GEMMs (K=Cin*4, half-res grid = Hin)
    auto convT = [&](const u16* Wp, const u16* X, int Cin, int Hin,
                     int bnIdx, u16* outb) {
        for (int p = 0; p < 4; p++) {
            ConvDesc cd{};
            cd.W = Wp + (size_t)p * 256 * (Cin * 4);
            cd.X = X; cd.M = 256; cd.K = Cin * 4; cd.Cin = Cin; cd.ks = 4;
            cd.stride = 2; cd.pad = 1; cd.NB = NB;
            cd.Hin = Hin; cd.Win = Hin;
            int wsh = __builtin_ctz(Hin);       // half-res output grid == Hin
            cd.wShift = wsh; cd.hwShift = 2 * wsh;
            cd.cinShift = __builtin_ctz(Cin);
            cd.parity = 1; cd.phH = p >> 1; cd.phW = p & 1;
            cd.in_relu = 0; cd.out_relu = 1;
            cd.has_bn = 1;
            cd.g = FP(bnIdx); cd.b = FP(bnIdx + 1);
            cd.m = FP(bnIdx + 2); cd.v = FP(bnIdx + 3);
            cd.has_bias = 0; cd.has_resid = 0; cd.logits = 0;
            cd.outb = outb; cd.outf = nullptr;
            int cols = NB * Hin * Hin;          // half-res col count
            dim3 grid(cols / NT, cd.M / MT);
            conv_wmma<<<grid, 256, 0, stream>>>(cd);
        }
    };

    float* out = (float*)d_out;
    const size_t LOGITS = (size_t)16 * 3 * 64 * 64 * 256;

    // ---------------- encoder ----------------
    conv(wl0,  xact, 256,   48,   3, 4, 2, 1, 64, 32, 0, E_BN1, -1, nullptr, 1, h1, nullptr, 0);
    conv(wl1,  h1,   256, 4096, 256, 4, 2, 1, 32, 16, 0, E_BN2, -1, nullptr, 0, h2, nullptr, 0);
    conv(wer1a, h2,  256, 2304, 256, 3, 1, 1, 16, 16, 1, E_R1_BN1, -1, nullptr, 1, tb, nullptr, 0);
    conv(wer1b, tb,  256,  256, 256, 1, 1, 0, 16, 16, 0, E_R1_BN2, -1, h2, 0, h2, nullptr, 0);
    conv(wer2a, h2,  256, 2304, 256, 3, 1, 1, 16, 16, 1, E_R2_BN1, -1, nullptr, 1, tb, nullptr, 0);
    conv(wer2b, tb,  256,  256, 256, 1, 1, 0, 16, 16, 0, E_R2_BN2, -1, h2, 0, h2, nullptr, 0);
    conv(wl4,  h2,   512,  256, 256, 1, 1, 0, 16, 16, 0, -1, E_B3, nullptr, 0, z, nullptr, 0);

    // ---------------- codebook ----------------
    {
        dim3 grid(4096 / 16, 8);
        codebook_argmin<<<grid, 32, 0, stream>>>(embB, e2, z, idx, hist, 4096);
        quantize_gather<<<8192, 256, 0, stream>>>(FP(EMB), idx, z, qv, part);
    }

    // ---------------- decoder ----------------
    conv(wd1,  qv,   256,  512, 512, 1, 1, 0, 16, 16, 0, D_BN1, -1, nullptr, 0, dd1, nullptr, 0);
    conv(wdr1a, dd1, 256, 2304, 256, 3, 1, 1, 16, 16, 1, D_R1_BN1, -1, nullptr, 1, tb, nullptr, 0);
    conv(wdr1b, tb,  256,  256, 256, 1, 1, 0, 16, 16, 0, D_R1_BN2, -1, dd1, 0, dd1, nullptr, 0);
    conv(wdr2a, dd1, 256, 2304, 256, 3, 1, 1, 16, 16, 1, D_R2_BN1, -1, nullptr, 1, tb, nullptr, 0);
    conv(wdr2b, tb,  256,  256, 256, 1, 1, 0, 16, 16, 0, D_R2_BN2, -1, dd1, 0, dd1, nullptr, 0);
    convT(wt1, dd1, 256, 16, D_BN2X, u1);    // -> 16x(32x32), relu(bn)
    convT(wt2, u1,  256, 32, D_BN3,  u2);    // -> 16x(64x64), relu(bn)
    conv(wd4,  u2,   768,  256, 256, 1, 1, 0, 64, 64, 0, -1, D_B4, nullptr, 0, nullptr, out, 1);

    finalize<<<1, 32, 0, stream>>>(hist, part, 8192, out, LOGITS);
}
